// Attention_57827439673540
// MI455X (gfx1250) — matrix operand order repaired and checked by reading, not compiled, not run on hardware
//
#include <hip/hip_runtime.h>
#include <hip/hip_bf16.h>

// ---------- types ----------
typedef __attribute__((ext_vector_type(16))) __bf16 v16bf;
typedef __attribute__((ext_vector_type(8)))  __bf16 v8bf;
typedef __attribute__((ext_vector_type(4)))  __bf16 v4bf;
typedef __attribute__((ext_vector_type(8)))  float  v8f;

#define BATCH 4
#define SEQ   2048
#define DIMM  1024
#define HEADS 16
#define DHEAD 64
#define NQKV  3072   // 3 * 1024

static __device__ __forceinline__ __bf16 f2bf(float f) {
    unsigned u = __builtin_bit_cast(unsigned, f);
    unsigned r = (u + 0x7fffu + ((u >> 16) & 1u)) >> 16;   // RNE
    unsigned short s = (unsigned short)r;
    return __builtin_bit_cast(__bf16, s);
}

static __device__ __forceinline__ v16bf cat8(v8bf lo, v8bf hi) {
    return __builtin_shufflevector(lo, hi, 0,1,2,3,4,5,6,7,8,9,10,11,12,13,14,15);
}

#define WMMA_BF16(A, B, C) \
    __builtin_amdgcn_wmma_f32_16x16x32_bf16(false, (A), false, (B), (short)0, (C), false, false)

// =====================================================================
// Kernel 1: QKV projection.  x[8192,1024] (f32) @ w_qkv[1024,3072] (f32)
// -> Q[bh][n][64], K[bh][n][64], Vt[bh][d][n]  (all bf16)
// Block: 256 threads (8 waves), tile 128(M) x 128(N), K-step 32.
// =====================================================================
__global__ __launch_bounds__(256) void qkv_gemm(
        const float* __restrict__ x, const float* __restrict__ w,
        __bf16* __restrict__ Q, __bf16* __restrict__ K, __bf16* __restrict__ Vt) {
    __shared__ __align__(32) __bf16 Asm[128 * 32];   // [row][k], row-major
    __shared__ __align__(32) __bf16 Bsm[128 * 32];   // [col][k], K-contig (transposed)
    const int tid  = threadIdx.x;
    const int lane = tid & 31;
    const int wave = tid >> 5;
    const int wm = wave & 3;        // wave row: 4 x 32 rows
    const int wn = wave >> 2;       // wave col: 2 x 64 cols
    const int bm = blockIdx.y;      // 64 blocks of 128 rows
    const int bn = blockIdx.x;      // 24 blocks of 128 cols

    v8f acc[2][4];
    #pragma unroll
    for (int i = 0; i < 2; ++i)
        #pragma unroll
        for (int j = 0; j < 4; ++j)
            #pragma unroll
            for (int e = 0; e < 8; ++e) acc[i][j][e] = 0.0f;

    for (int kb = 0; kb < 32; ++kb) {
        __syncthreads();
        // ---- stage A tile: 128x32 f32 -> bf16 LDS (row-major) ----
        #pragma unroll
        for (int p = 0; p < 4; ++p) {
            int row = (tid >> 3) + p * 32;
            int kc  = (tid & 7) * 4;
            const float4 f = *reinterpret_cast<const float4*>(
                x + (size_t)(bm * 128 + row) * DIMM + kb * 32 + kc);
            v4bf b; b[0] = f2bf(f.x); b[1] = f2bf(f.y); b[2] = f2bf(f.z); b[3] = f2bf(f.w);
            *reinterpret_cast<v4bf*>(&Asm[row * 32 + kc]) = b;
        }
        // ---- stage B tile: 32x128 f32 -> bf16 LDS transposed [col][k] ----
        #pragma unroll
        for (int p = 0; p < 4; ++p) {
            int kr = (tid >> 5) + p * 8;
            int nc = (tid & 31) * 4;
            const float4 f = *reinterpret_cast<const float4*>(
                w + (size_t)(kb * 32 + kr) * NQKV + bn * 128 + nc);
            Bsm[(nc + 0) * 32 + kr] = f2bf(f.x);
            Bsm[(nc + 1) * 32 + kr] = f2bf(f.y);
            Bsm[(nc + 2) * 32 + kr] = f2bf(f.z);
            Bsm[(nc + 3) * 32 + kr] = f2bf(f.w);
        }
        if (kb + 1 < 32) {
            __builtin_prefetch(x + (size_t)(bm * 128 + (tid >> 3)) * DIMM + (kb + 1) * 32, 0, 1);
            __builtin_prefetch(w + (size_t)((kb + 1) * 32 + (tid >> 5)) * NQKV + bn * 128, 0, 1);
        }
        __syncthreads();
        // ---- compute: 2x4 WMMA tiles per wave ----
        v16bf bf[4];
        #pragma unroll
        for (int tj = 0; tj < 4; ++tj) {
            int col = wn * 64 + tj * 16 + (lane & 15);
            bf[tj] = *reinterpret_cast<const v16bf*>(&Bsm[col * 32 + (lane >> 4) * 16]);
        }
        #pragma unroll
        for (int ti = 0; ti < 2; ++ti) {
            int row = wm * 32 + ti * 16 + (lane & 15);
            v8bf lo = *reinterpret_cast<const v8bf*>(&Asm[row * 32 + (lane >> 4) * 16]);
            v8bf hi = *reinterpret_cast<const v8bf*>(&Asm[row * 32 + 8 + (lane >> 4) * 16]);
            v16bf a = cat8(lo, hi);
            #pragma unroll
            for (int tj = 0; tj < 4; ++tj)
                acc[ti][tj] = WMMA_BF16(a, bf[tj], acc[ti][tj]);
        }
    }

    // ---- epilogue: scatter to Q / K (row-major [bh][n][d]) and Vt ([bh][d][n]) ----
    #pragma unroll
    for (int ti = 0; ti < 2; ++ti) {
        #pragma unroll
        for (int tj = 0; tj < 4; ++tj) {
            #pragma unroll
            for (int r = 0; r < 8; ++r) {
                int row = bm * 128 + wm * 32 + ti * 16 + r + ((lane >> 4) * 8);
                int col = bn * 128 + wn * 64 + tj * 16 + (lane & 15);
                int batch = row >> 11, npos = row & 2047;
                int chunk = col >> 10, inner = col & 1023;
                int h = inner >> 6,  d = inner & 63;
                size_t bh = (size_t)(batch * HEADS + h);
                __bf16 bv = f2bf(acc[ti][tj][r]);
                if (chunk == 0)      Q [(bh * SEQ + npos) * DHEAD + d] = bv;
                else if (chunk == 1) K [(bh * SEQ + npos) * DHEAD + d] = bv;
                else                 Vt[(bh * DHEAD + d) * SEQ + npos] = bv;
            }
        }
    }
}

// =====================================================================
// Kernel 2: flash attention.  One workgroup = 4 waves = 64 queries of one
// (batch,head); each wave owns 16 query rows, streams keys in blocks of 32.
// O written as [b][n][h*64+d] bf16 so the out-proj GEMM reads row-major.
// =====================================================================
__global__ __launch_bounds__(128) void flash_attn(
        const __bf16* __restrict__ Q, const __bf16* __restrict__ K,
        const __bf16* __restrict__ Vt, __bf16* __restrict__ O) {
    __shared__ __align__(32) __bf16 Psm[4][16 * 32];   // per-wave P tile, row-major
    const int tid  = threadIdx.x;
    const int lane = tid & 31;
    const int wave = tid >> 5;
    const int bh   = blockIdx.y;                 // 0..63
    const int q0   = blockIdx.x * 64 + wave * 16;
    const size_t base = (size_t)bh * SEQ * DHEAD;

    // Q A-fragments (16x32 each, d-chunks 0 and 1), loaded once
    v16bf qa[2];
    {
        const __bf16* qrow = Q + base + (size_t)(q0 + (lane & 15)) * DHEAD;
        #pragma unroll
        for (int c = 0; c < 2; ++c) {
            v8bf lo = *reinterpret_cast<const v8bf*>(qrow + c * 32 + ((lane >> 4) * 16));
            v8bf hi = *reinterpret_cast<const v8bf*>(qrow + c * 32 + 8 + ((lane >> 4) * 16));
            qa[c] = cat8(lo, hi);
        }
    }

    v8f o[4];
    #pragma unroll
    for (int tj = 0; tj < 4; ++tj)
        #pragma unroll
        for (int e = 0; e < 8; ++e) o[tj][e] = 0.0f;
    float m[8], l[8];
    #pragma unroll
    for (int r = 0; r < 8; ++r) { m[r] = -3.0e38f; l[r] = 0.0f; }

    for (int kb = 0; kb < SEQ; kb += 32) {
        // ---- S = Q * K^T for 32 keys (two 16-wide C tiles, K-dim = 64) ----
        v8f s[2];
        #pragma unroll
        for (int t = 0; t < 2; ++t)
            #pragma unroll
            for (int e = 0; e < 8; ++e) s[t][e] = 0.0f;
        #pragma unroll
        for (int t = 0; t < 2; ++t) {
            #pragma unroll
            for (int c = 0; c < 2; ++c) {
                const __bf16* kp = K + base
                    + (size_t)(kb + t * 16 + (lane & 15)) * DHEAD + c * 32 + (lane >> 4) * 16;
                v16bf kf = *reinterpret_cast<const v16bf*>(kp);
                s[t] = WMMA_BF16(qa[c], kf, s[t]);
            }
        }
        // ---- online softmax (rows live across 16-lane halves per C layout) ----
        #pragma unroll
        for (int r = 0; r < 8; ++r) {
            float a0 = s[0][r] * 0.125f;          // 1/sqrt(64)
            float a1 = s[1][r] * 0.125f;
            float mx = fmaxf(a0, a1);
            mx = fmaxf(mx, __shfl_xor(mx, 1, 16));
            mx = fmaxf(mx, __shfl_xor(mx, 2, 16));
            mx = fmaxf(mx, __shfl_xor(mx, 4, 16));
            mx = fmaxf(mx, __shfl_xor(mx, 8, 16));
            float mn = fmaxf(m[r], mx);
            float al = __expf(m[r] - mn);
            float p0 = __expf(a0 - mn);
            float p1 = __expf(a1 - mn);
            float rs = p0 + p1;
            rs += __shfl_xor(rs, 1, 16);
            rs += __shfl_xor(rs, 2, 16);
            rs += __shfl_xor(rs, 4, 16);
            rs += __shfl_xor(rs, 8, 16);
            l[r] = l[r] * al + rs;
            m[r] = mn;
            #pragma unroll
            for (int tj = 0; tj < 4; ++tj) o[tj][r] *= al;
            // stash P (bf16) in per-wave LDS, row-major 16x32
            int prow = r + ((lane >> 4) * 8);
            Psm[wave][prow * 32 + (lane & 15)]      = f2bf(p0);
            Psm[wave][prow * 32 + 16 + (lane & 15)] = f2bf(p1);
        }
        // LDS ops are in-order per wave; explicit split-counter wait for safety
        asm volatile("s_wait_dscnt 0" ::: "memory");
        // reload P as A-fragment (16x32)
        v8bf plo = *reinterpret_cast<const v8bf*>(&Psm[wave][(lane & 15) * 32 + ((lane >> 4) * 16)]);
        v8bf phi = *reinterpret_cast<const v8bf*>(&Psm[wave][(lane & 15) * 32 + 8 + ((lane >> 4) * 16)]);
        v16bf pa = cat8(plo, phi);
        // ---- O += P * V  (V^T stored so K-dim (keys) is contiguous) ----
        #pragma unroll
        for (int tj = 0; tj < 4; ++tj) {
            const __bf16* vp = Vt + base
                + (size_t)(tj * 16 + (lane & 15)) * SEQ + kb + (lane >> 4) * 16;
            v16bf vf = *reinterpret_cast<const v16bf*>(vp);
            o[tj] = WMMA_BF16(pa, vf, o[tj]);
        }
    }

    // ---- epilogue: normalize and write [b][n][h*64+d] ----
    int batch = bh >> 4, h = bh & 15;
    #pragma unroll
    for (int r = 0; r < 8; ++r) {
        float inv = 1.0f / l[r];
        int qrow = q0 + r + ((lane >> 4) * 8);
        #pragma unroll
        for (int tj = 0; tj < 4; ++tj) {
            int d = tj * 16 + (lane & 15);
            O[(size_t)(batch * SEQ + qrow) * DIMM + h * DHEAD + d] = f2bf(o[tj][r] * inv);
        }
    }
}

// =====================================================================
// Kernel 3: output projection.  O[8192,1024] (bf16) @ w_out[1024,1024] (f32)
// + b_out -> out (f32).  Same tiling as kernel 1.
// =====================================================================
__global__ __launch_bounds__(256) void out_gemm(
        const __bf16* __restrict__ A, const float* __restrict__ w,
        const float* __restrict__ bias, float* __restrict__ out) {
    __shared__ __align__(32) __bf16 Asm[128 * 32];
    __shared__ __align__(32) __bf16 Bsm[128 * 32];
    const int tid  = threadIdx.x;
    const int lane = tid & 31;
    const int wave = tid >> 5;
    const int wm = wave & 3, wn = wave >> 2;
    const int bm = blockIdx.y, bn = blockIdx.x;

    v8f acc[2][4];
    #pragma unroll
    for (int i = 0; i < 2; ++i)
        #pragma unroll
        for (int j = 0; j < 4; ++j)
            #pragma unroll
            for (int e = 0; e < 8; ++e) acc[i][j][e] = 0.0f;

    for (int kb = 0; kb < 32; ++kb) {
        __syncthreads();
        #pragma unroll
        for (int p = 0; p < 4; ++p) {
            int row = (tid >> 3) + p * 32;
            int kc  = (tid & 7) * 4;
            v4bf av = *reinterpret_cast<const v4bf*>(
                A + (size_t)(bm * 128 + row) * DIMM + kb * 32 + kc);
            *reinterpret_cast<v4bf*>(&Asm[row * 32 + kc]) = av;
        }
        #pragma unroll
        for (int p = 0; p < 4; ++p) {
            int kr = (tid >> 5) + p * 8;
            int nc = (tid & 31) * 4;
            const float4 f = *reinterpret_cast<const float4*>(
                w + (size_t)(kb * 32 + kr) * DIMM + bn * 128 + nc);
            Bsm[(nc + 0) * 32 + kr] = f2bf(f.x);
            Bsm[(nc + 1) * 32 + kr] = f2bf(f.y);
            Bsm[(nc + 2) * 32 + kr] = f2bf(f.z);
            Bsm[(nc + 3) * 32 + kr] = f2bf(f.w);
        }
        if (kb + 1 < 32) {
            __builtin_prefetch(A + (size_t)(bm * 128 + (tid >> 3)) * DIMM + (kb + 1) * 32, 0, 1);
            __builtin_prefetch(w + (size_t)((kb + 1) * 32 + (tid >> 5)) * DIMM + bn * 128, 0, 1);
        }
        __syncthreads();
        v16bf bf[4];
        #pragma unroll
        for (int tj = 0; tj < 4; ++tj) {
            int col = wn * 64 + tj * 16 + (lane & 15);
            bf[tj] = *reinterpret_cast<const v16bf*>(&Bsm[col * 32 + (lane >> 4) * 16]);
        }
        #pragma unroll
        for (int ti = 0; ti < 2; ++ti) {
            int row = wm * 32 + ti * 16 + (lane & 15);
            v8bf lo = *reinterpret_cast<const v8bf*>(&Asm[row * 32 + (lane >> 4) * 16]);
            v8bf hi = *reinterpret_cast<const v8bf*>(&Asm[row * 32 + 8 + (lane >> 4) * 16]);
            v16bf a = cat8(lo, hi);
            #pragma unroll
            for (int tj = 0; tj < 4; ++tj)
                acc[ti][tj] = WMMA_BF16(a, bf[tj], acc[ti][tj]);
        }
    }

    #pragma unroll
    for (int ti = 0; ti < 2; ++ti) {
        #pragma unroll
        for (int tj = 0; tj < 4; ++tj) {
            #pragma unroll
            for (int r = 0; r < 8; ++r) {
                int row = bm * 128 + wm * 32 + ti * 16 + r + ((lane >> 4) * 8);
                int col = bn * 128 + wn * 64 + tj * 16 + (lane & 15);
                out[(size_t)row * DIMM + col] = acc[ti][tj][r] + bias[col];
            }
        }
    }
}

// =====================================================================
extern "C" void kernel_launch(void* const* d_in, const int* in_sizes, int n_in,
                              void* d_out, int out_size, void* d_ws, size_t ws_size,
                              hipStream_t stream) {
    (void)in_sizes; (void)n_in; (void)out_size; (void)ws_size;
    const float* x     = (const float*)d_in[0];
    const float* w_qkv = (const float*)d_in[1];
    const float* w_out = (const float*)d_in[2];
    const float* b_out = (const float*)d_in[3];
    float* out = (float*)d_out;

    const size_t QKV_ELEMS = (size_t)BATCH * HEADS * SEQ * DHEAD;   // 8,388,608
    __bf16* Q  = (__bf16*)d_ws;
    __bf16* K  = Q  + QKV_ELEMS;
    __bf16* Vt = K  + QKV_ELEMS;
    __bf16* O  = Vt + QKV_ELEMS;   // total 64 MB bf16 workspace

    qkv_gemm  <<<dim3(NQKV / 128, (BATCH * SEQ) / 128), 256, 0, stream>>>(x, w_qkv, Q, K, Vt);
    flash_attn<<<dim3(SEQ / 64, BATCH * HEADS),        128, 0, stream>>>(Q, K, Vt, O);
    out_gemm  <<<dim3(DIMM / 128, (BATCH * SEQ) / 128), 256, 0, stream>>>(O, w_out, b_out, out);
}
